// TopKGate_17806934409743
// MI455X (gfx1250) — matrix operand order repaired and checked by reading, not compiled, not run on hardware
//
#include <hip/hip_runtime.h>

#define T_TOKENS 8192
#define MODEL_D  4096
#define NEXP     64

#define OFF_LOGITS 0
#define OFF_IDX    (T_TOKENS * NEXP)            // 524288
#define OFF_LOC    (OFF_IDX + 2 * T_TOKENS)     // 540672
#define OFF_GATE   (OFF_LOC + 2 * T_TOKENS)     // 557056
#define OFF_LAUX   (OFF_GATE + 2 * T_TOKENS)    // 573440

typedef _Float16 v8h  __attribute__((ext_vector_type(8)));
typedef _Float16 v16h __attribute__((ext_vector_type(16)));
typedef float    v8f  __attribute__((ext_vector_type(8)));

// ---------------------------------------------------------------------------
// Kernel 1: logits = x @ w_gate  (fp32 in, f16 WMMA compute, fp32 accumulate)
// Block = 256 threads = 8 waves; block tile = 32 rows x 64 experts.
// Wave (w>>2) selects M-subtile (0/16), (w&3) selects N-subtile (0/16/32/48).
// ---------------------------------------------------------------------------
__global__ __launch_bounds__(256) void moe_gemm_logits(
    const float* __restrict__ x, const float* __restrict__ wg,
    float* __restrict__ logits) {
  // padded strides (40 halves = 80 B) -> 16B-aligned, bank-conflict-friendly
  __shared__ __align__(16) _Float16 As[32 * 40];  // [row][k]
  __shared__ __align__(16) _Float16 Bs[64 * 40];  // [expert][k] (col-major B)

  const int tid   = threadIdx.x;
  const int wave  = tid >> 5;
  const int lane  = tid & 31;
  const int l15   = lane & 15;
  const int hi    = lane >> 4;
  const int m0    = blockIdx.x * 32;
  const int m_off = (wave >> 2) * 16;
  const int n_off = (wave & 3) * 16;

  // A staging: thread -> (row 0..31, 4 consecutive k)
  const int ar = tid >> 3;
  const int ak = (tid & 7) * 4;
  // B staging: thread -> (k 0..31, 8 consecutive experts)
  const int bk = tid & 31;
  const int bn = (tid >> 5) * 8;

  v8f acc = {};

  for (int k0 = 0; k0 < MODEL_D; k0 += 32) {
    // ---- stage A chunk: 32 rows x 32 k of x, fp32 -> f16 ----
    const float4 xa = *(const float4*)(x + (size_t)(m0 + ar) * MODEL_D + k0 + ak);
    As[ar * 40 + ak + 0] = (_Float16)xa.x;
    As[ar * 40 + ak + 1] = (_Float16)xa.y;
    As[ar * 40 + ak + 2] = (_Float16)xa.z;
    As[ar * 40 + ak + 3] = (_Float16)xa.w;

    // ---- stage B chunk: 32 k x 64 experts of w_gate, transpose to [n][k] ----
    const float4 w0 = *(const float4*)(wg + (size_t)(k0 + bk) * NEXP + bn);
    const float4 w1 = *(const float4*)(wg + (size_t)(k0 + bk) * NEXP + bn + 4);
    Bs[(bn + 0) * 40 + bk] = (_Float16)w0.x;
    Bs[(bn + 1) * 40 + bk] = (_Float16)w0.y;
    Bs[(bn + 2) * 40 + bk] = (_Float16)w0.z;
    Bs[(bn + 3) * 40 + bk] = (_Float16)w0.w;
    Bs[(bn + 4) * 40 + bk] = (_Float16)w1.x;
    Bs[(bn + 5) * 40 + bk] = (_Float16)w1.y;
    Bs[(bn + 6) * 40 + bk] = (_Float16)w1.z;
    Bs[(bn + 7) * 40 + bk] = (_Float16)w1.w;

    if (k0 + 32 < MODEL_D)  // hint next A chunk -> global_prefetch_b8
      __builtin_prefetch(x + (size_t)(m0 + ar) * MODEL_D + k0 + 32 + ak, 0, 1);

    __syncthreads();

    // ---- build WMMA fragments per ISA lane layout ----
    // A 16x32 f16: lanes 0-15 row=l15 hold K{0..7,16..23}; lanes 16-31 K{8..15,24..31}
    const _Float16* arow = &As[(m_off + l15) * 40];
    const v8h a0 = *(const v8h*)(arow + (hi ? 8 : 0));
    const v8h a1 = *(const v8h*)(arow + (hi ? 24 : 16));
    // B 32x16 f16: lanes 0-15 col=l15 hold K{0..7,16..23}; lanes 16-31 K{8..15,24..31}
    const _Float16* bcol = &Bs[(n_off + l15) * 40 + hi * 8];
    const v8h b0 = *(const v8h*)(bcol);
    const v8h b1 = *(const v8h*)(bcol + 16);

    v16h a, b;
#pragma unroll
    for (int i = 0; i < 8; ++i) {
      a[i] = a0[i]; a[i + 8] = a1[i];
      b[i] = b0[i]; b[i + 8] = b1[i];
    }

    acc = __builtin_amdgcn_wmma_f32_16x16x32_f16(
        /*neg_a=*/false, a, /*neg_b=*/false, b,
        /*c_mod=*/(short)0, acc, /*reuse_a=*/false, /*reuse_b=*/false);

    __syncthreads();
  }

  // C/D layout: lanes 0-15 -> M=i, N=l15 ; lanes 16-31 -> M=i+8, N=l15
#pragma unroll
  for (int i = 0; i < 8; ++i) {
    const int m = m0 + m_off + i + 8 * hi;
    logits[(size_t)m * NEXP + n_off + l15] = acc[i];
  }
}

// ---------------------------------------------------------------------------
// Kernel 2: per-token softmax, top-2, normalized gates; stash rowmax/rowsum.
// ---------------------------------------------------------------------------
__global__ __launch_bounds__(256) void moe_route(
    const float* __restrict__ logits, float* __restrict__ out,
    float* __restrict__ rowstats) {
  const int t = blockIdx.x * 256 + threadIdx.x;
  const float* row = logits + (size_t)t * NEXP;

  float mx = row[0];
#pragma unroll 8
  for (int j = 1; j < NEXP; ++j) mx = fmaxf(mx, row[j]);

  float sum = 0.0f;
  float b0 = -3.0e38f, b1 = -3.0e38f;
  int i0 = 0, i1 = 0;
#pragma unroll 8
  for (int j = 0; j < NEXP; ++j) {
    const float v = row[j];
    sum += __expf(v - mx);
    if (v > b0)      { b1 = b0; i1 = i0; b0 = v; i0 = j; }
    else if (v > b1) { b1 = v; i1 = j; }
  }
  const float inv = 1.0f / sum;
  const float g0 = __expf(b0 - mx) * inv;
  const float g1 = __expf(b1 - mx) * inv;
  const float den = fmaxf(g0 + g1, 1.1920929e-07f);

  out[OFF_IDX + t]             = (float)i0;
  out[OFF_IDX + T_TOKENS + t]  = (float)i1;
  out[OFF_GATE + t]            = g0 / den;
  out[OFF_GATE + T_TOKENS + t] = g1 / den;
  rowstats[2 * t]     = mx;
  rowstats[2 * t + 1] = sum;
}

// ---------------------------------------------------------------------------
// Kernel 3: token-ordered per-expert prefix -> locations.  One block, 1024 thr.
// counts[] carries across chunks and across k=0 -> k=1 (== acc_base).
// ---------------------------------------------------------------------------
__global__ __launch_bounds__(1024) void moe_locations(
    const float* __restrict__ idx_in, float* __restrict__ loc_out) {
  __shared__ int counts[NEXP];
  __shared__ int experts[1024];
  __shared__ int hist[32][NEXP];

  const int tid  = threadIdx.x;
  const int wave = tid >> 5;
  const int lane = tid & 31;

  if (tid < NEXP) counts[tid] = 0;
  __syncthreads();

  for (int k = 0; k < 2; ++k) {
    for (int c = 0; c < T_TOKENS / 1024; ++c) {
      const int t = c * 1024 + tid;
      const int e = (int)idx_in[k * T_TOKENS + t];
      experts[tid] = e;
      hist[tid >> 6][tid & 63] = 0;
      if (tid < 1024) hist[16 + (tid >> 6)][tid & 63] = 0;
      __syncthreads();

      atomicAdd(&hist[wave][e], 1);
      __syncthreads();

      int base = counts[e];
      for (int w = 0; w < wave; ++w) base += hist[w][e];
      const int wbase = wave * 32;
      for (int j = 0; j < lane; ++j) base += (experts[wbase + j] == e) ? 1 : 0;
      loc_out[k * T_TOKENS + t] = (float)base;
      __syncthreads();

      if (tid < NEXP) {
        int s = counts[tid];
        for (int w = 0; w < 32; ++w) s += hist[w][tid];
        counts[tid] = s;
      }
      __syncthreads();
    }
  }
}

// ---------------------------------------------------------------------------
// Kernel 4: per-expert me = mean softmax score, ce = mean top-1 mask.
// Fixed-order tree reduction -> deterministic.  64 blocks x 256 threads.
// ---------------------------------------------------------------------------
__global__ __launch_bounds__(256) void moe_mece(
    const float* __restrict__ logits, const float* __restrict__ rowstats,
    const float* __restrict__ idx0, float* __restrict__ mece) {
  const int e = blockIdx.x;
  const int tid = threadIdx.x;
  float me = 0.0f, ce = 0.0f;
  for (int t = tid; t < T_TOKENS; t += 256) {
    const float sc = __expf(logits[(size_t)t * NEXP + e] - rowstats[2 * t]) /
                     rowstats[2 * t + 1];
    me += sc;
    ce += (idx0[t] == (float)e) ? 1.0f : 0.0f;
  }
  __shared__ float sm[256], sc_[256];
  sm[tid] = me; sc_[tid] = ce;
  __syncthreads();
  for (int s = 128; s > 0; s >>= 1) {
    if (tid < s) { sm[tid] += sm[tid + s]; sc_[tid] += sc_[tid + s]; }
    __syncthreads();
  }
  if (tid == 0)
    mece[e] = (sm[0] / (float)T_TOKENS) * (sc_[0] / (float)T_TOKENS);
}

__global__ __launch_bounds__(64) void moe_laux(
    const float* __restrict__ mece, float* __restrict__ out) {
  __shared__ float s[NEXP];
  const int tid = threadIdx.x;
  s[tid] = mece[tid];
  __syncthreads();
  for (int st = 32; st > 0; st >>= 1) {
    if (tid < st) s[tid] += s[tid + st];
    __syncthreads();
  }
  if (tid == 0) {
    out[OFF_LAUX]     = s[0] * (float)NEXP;  // l_aux
    out[OFF_LAUX + 1] = 256.0f;              // capacity = 2*ceil(8192/64)
  }
}

// ---------------------------------------------------------------------------
extern "C" void kernel_launch(void* const* d_in, const int* in_sizes, int n_in,
                              void* d_out, int out_size, void* d_ws, size_t ws_size,
                              hipStream_t stream) {
  (void)in_sizes; (void)n_in; (void)out_size; (void)ws_size;
  const float* x  = (const float*)d_in[0];
  const float* wg = (const float*)d_in[1];
  float* out      = (float*)d_out;
  float* rowstats = (float*)d_ws;              // [T][2] : rowmax, rowsum
  float* mece     = rowstats + 2 * T_TOKENS;   // [64]

  moe_gemm_logits<<<T_TOKENS / 32, 256, 0, stream>>>(x, wg, out + OFF_LOGITS);
  moe_route<<<T_TOKENS / 256, 256, 0, stream>>>(out + OFF_LOGITS, out, rowstats);
  moe_locations<<<1, 1024, 0, stream>>>(out + OFF_IDX, out + OFF_LOC);
  moe_mece<<<NEXP, 256, 0, stream>>>(out + OFF_LOGITS, rowstats, out + OFF_IDX, mece);
  moe_laux<<<1, 64, 0, stream>>>(mece, out + OFF_LAUX);
}
